// GRU_75505525064434
// MI455X (gfx1250) — compile-verified, matrix operand order checked by reading, not run on hardware
//
#include <hip/hip_runtime.h>
#include <hip/hip_bf16.h>
#include <stdint.h>

// ---------------------------------------------------------------------------
// CDNA5 (gfx1250) bidirectional GRU. wave32, WMMA 16x16x32 bf16 -> f32.
//
// Input order assumption (setup_inputs() insertion order, flattened):
//  0:x [B*T] i32   1:l [B] i32   2:emb [V*E] f32
//  3..6   gru[0].fwd  W_ih[768*128], W_hh[768*256], b_ih[768], b_hh[768]
//  7..10  gru[0].bwd  (same shapes)
//  11..14 gru[1].fwd  W_ih[768*512], W_hh[768*256], b_ih[768], b_hh[768]
//  15..18 gru[1].bwd
//  19:w_out [512]   20:b_out [1]
// Output: [B]=128 f32.
// ---------------------------------------------------------------------------

#define Bv 128
#define Tv 1024
#define Ev 128
#define Hv 256
#define G3H 768           // 3*H
#define MROWS (Bv * Tv)   // 131072

typedef __attribute__((ext_vector_type(16))) __bf16 bf16x16;
typedef __attribute__((ext_vector_type(8)))  float  f32x8;

union Frag { bf16x16 v; unsigned u[8]; };

// 16-bit A/B matrix VGPR layout (CDNA5 ISA §7.12.2):
// lanes 0-15 hold K pairs {0,1},{2,3},{4,5},{6,7} in v0..3 and {16..23} in v4..7;
// lanes 16-31 hold the +8 shifted ranges. Same pattern for B with N in lanes.
// Contiguity per half => compiler fuses the 8 dword loads into 2x b128.
__device__ __forceinline__ int wmma_koff(int v, int laneHi) {
    return ((v < 4) ? (2 * v) : (16 + 2 * (v - 4))) + (laneHi ? 8 : 0);
}

__device__ __forceinline__ float sigmoidf_(float x) {
    return 1.0f / (1.0f + __expf(-x));
}

// ---------------------------------------------------------------------------
// fp32 -> bf16 elementwise (weights)
// ---------------------------------------------------------------------------
__global__ void f32_to_bf16_kernel(const float* __restrict__ in,
                                   __bf16* __restrict__ out, int n) {
    int i = blockIdx.x * blockDim.x + threadIdx.x;
    if (i < n) out[i] = (__bf16)in[i];
}

// ---------------------------------------------------------------------------
// Embedding gather -> bf16 [B*T, E], float4-vectorized (8B bf16 stores)
// ---------------------------------------------------------------------------
__global__ void embed_bf16_kernel(const int* __restrict__ x,
                                  const float* __restrict__ emb,
                                  __bf16* __restrict__ X0) {
    size_t q = (size_t)blockIdx.x * blockDim.x + threadIdx.x;  // over MROWS*32
    if (q >= (size_t)MROWS * (Ev / 4)) return;
    const size_t bt = q >> 5;               // / 32 quads per row
    const int e = (int)(q & 31) * 4;
    const int row = x[bt];
    const float4 f = *(const float4*)(emb + (size_t)row * Ev + e);
    __bf16 o[4] = {(__bf16)f.x, (__bf16)f.y, (__bf16)f.z, (__bf16)f.w};
    *(uint2*)(X0 + (bt << 7) + e) = *(const uint2*)o;
}

// ---------------------------------------------------------------------------
// GX = A[M,K] * W[768,K]^T + bias  -> bf16 [M,768]
// Block: 256 thr = 8 waves; block tile 128(M) x 96(N); wave tile 16 x 96.
// K-loop of v_wmma_f32_16x16x32_bf16, A fragment reused across 6 N-tiles.
// GX is stream-out (read exactly once by the scan) -> non-temporal stores.
// ---------------------------------------------------------------------------
__global__ void __launch_bounds__(256)
gemm_gates_bf16(const __bf16* __restrict__ A,     // [M,K] row-major
                const __bf16* __restrict__ W,     // [768,K] row-major (B^T)
                const float*  __restrict__ bias,  // [768]
                __bf16* __restrict__ GX,          // [M,768]
                int K) {
    const int lane   = threadIdx.x & 31;
    const int wave   = threadIdx.x >> 5;   // 0..7
    const int laneHi = lane >> 4;
    const int nlo    = lane & 15;
    const int m0     = blockIdx.x * 128 + wave * 16;
    const int n0     = blockIdx.y * 96;

    const f32x8 zero = {};
    f32x8 acc[6];
#pragma unroll
    for (int i = 0; i < 6; ++i) acc[i] = zero;

    const __bf16* arow = A + (size_t)(m0 + nlo) * K;  // A: M in lanes 0..15
    for (int kk = 0; kk < K; kk += 32) {
        Frag a;
#pragma unroll
        for (int v = 0; v < 8; ++v)
            a.u[v] = *(const unsigned*)(arow + kk + wmma_koff(v, laneHi));
#pragma unroll
        for (int nt = 0; nt < 6; ++nt) {
            const __bf16* wrow = W + (size_t)(n0 + nt * 16 + nlo) * K;
            Frag b;
#pragma unroll
            for (int v = 0; v < 8; ++v)
                b.u[v] = *(const unsigned*)(wrow + kk + wmma_koff(v, laneHi));
            acc[nt] = __builtin_amdgcn_wmma_f32_16x16x32_bf16(
                false, a.v, false, b.v, (short)0, acc[nt], false, false);
        }
    }
    // C/D layout: vgpr i -> M = i + 8*laneHi, N = lane&15
#pragma unroll
    for (int nt = 0; nt < 6; ++nt) {
        const int col = n0 + nt * 16 + nlo;
        const float bv = bias[col];
#pragma unroll
        for (int i = 0; i < 8; ++i) {
            const int row = m0 + i + (laneHi << 3);
            __builtin_nontemporal_store((__bf16)(acc[nt][i] + bv),
                                        &GX[(size_t)row * G3H + col]);
        }
    }
}

// ---------------------------------------------------------------------------
// GRU scan. grid.x = 2 (0: fwd, 1: bwd). 1024 threads = 32 waves.
// Per step: hg = h[128,256] * Whh[768,256]^T via WMMA with h in LDS (bf16,
// row stride 264 to dodge bank conflicts). Each wave owns 4 combos
// (m-tile, 16-col group) covering gates r/z/n at identical (b,j) lanes,
// so the nonlinearity is lane-local. Two barriers per step.
// gx for the NEXT step is prefetched (global_prefetch) under the WMMA block
// so the HBM-latency of the 196KB/step gate stream is off the critical path;
// gx reads are non-temporal (single use, 402MB stream > 192MB L2).
// ---------------------------------------------------------------------------
__global__ void __launch_bounds__(1024)
gru_scan_kernel(const __bf16* __restrict__ gxF, const __bf16* __restrict__ gxB,
                const __bf16* __restrict__ WhhF, const __bf16* __restrict__ WhhB,
                const float* __restrict__ bhhF, const float* __restrict__ bhhB,
                const int* __restrict__ len,
                __bf16* __restrict__ outBuf,   // [B,T,512] (next-layer input)
                int writeOut,
                float* __restrict__ hFinF, float* __restrict__ hFinB) {
    const int dir = blockIdx.x;  // 0 fwd, 1 bwd
    const __bf16* gx  = dir ? gxB  : gxF;
    const __bf16* Whh = dir ? WhhB : WhhF;
    const float*  bhh = dir ? bhhB : bhhF;
    float* hFin = dir ? hFinB : hFinF;
    const int outOff = dir ? Hv : 0;

    const int HSTRIDE = Hv + 8;  // 264 bf16 per row
    extern __shared__ __align__(16) char smem[];
    __bf16* hsh = (__bf16*)smem;                       // [128][264]
    int*    lsh = (int*)(smem + (size_t)Bv * HSTRIDE * 2);

    for (int i = threadIdx.x; i < Bv * HSTRIDE; i += 1024) hsh[i] = (__bf16)0.0f;
    if (threadIdx.x < Bv) lsh[threadIdx.x] = len[threadIdx.x];
    __syncthreads();

    const int lane   = threadIdx.x & 31;
    const int wave   = threadIdx.x >> 5;  // 0..31
    const int laneHi = lane >> 4;
    const int nlo    = lane & 15;

    for (int s = 0; s < Tv; ++s) {
        const int t = dir ? (Tv - 1 - s) : s;

        // Prefetch next step's gate rows (12 x 128B lines per batch row).
        if (s + 1 < Tv) {
            const int tn = dir ? (t - 1) : (t + 1);
            for (int pi = threadIdx.x; pi < Bv * 12; pi += 1024) {
                const int prow = pi / 12, pline = pi % 12;
                __builtin_prefetch(
                    gx + ((size_t)prow * Tv + tn) * G3H + pline * 64, 0, 1);
            }
        }

        float hstage[4][8];
#pragma unroll
        for (int c = 0; c < 4; ++c) {
            const int combo = wave * 4 + c;        // 0..127
            const int m0 = (combo >> 4) * 16;      // batch tile
            const int j0 = (combo & 15) * 16;      // hidden-col group
            const f32x8 zero = {};
            f32x8 ar = zero, az = zero, an = zero;

            const __bf16* hrow = hsh + (size_t)(m0 + nlo) * HSTRIDE;
            const __bf16* wr = Whh + (size_t)(j0 + nlo) * Hv;
            const __bf16* wz = wr + (size_t)Hv * Hv;
            const __bf16* wn = wz + (size_t)Hv * Hv;
#pragma unroll
            for (int kk = 0; kk < Hv; kk += 32) {
                Frag a, fr, fz, fn;
#pragma unroll
                for (int v = 0; v < 8; ++v) {
                    const int ko = kk + wmma_koff(v, laneHi);
                    a.u[v]  = *(const unsigned*)(hrow + ko);   // ds_load
                    fr.u[v] = *(const unsigned*)(wr + ko);
                    fz.u[v] = *(const unsigned*)(wz + ko);
                    fn.u[v] = *(const unsigned*)(wn + ko);
                }
                ar = __builtin_amdgcn_wmma_f32_16x16x32_bf16(
                    false, a.v, false, fr.v, (short)0, ar, false, false);
                az = __builtin_amdgcn_wmma_f32_16x16x32_bf16(
                    false, a.v, false, fz.v, (short)0, az, false, false);
                an = __builtin_amdgcn_wmma_f32_16x16x32_bf16(
                    false, a.v, false, fn.v, (short)0, an, false, false);
            }
            const int col = j0 + nlo;
            const float br_ = bhh[col], bz_ = bhh[Hv + col], bn_ = bhh[2 * Hv + col];
#pragma unroll
            for (int i = 0; i < 8; ++i) {
                const int brow = m0 + i + (laneHi << 3);
                const size_t gxo = ((size_t)brow * Tv + t) * G3H;
                const float gr = (float)__builtin_nontemporal_load(&gx[gxo + col]);
                const float gz = (float)__builtin_nontemporal_load(&gx[gxo + Hv + col]);
                const float gn = (float)__builtin_nontemporal_load(&gx[gxo + 2 * Hv + col]);
                const float r  = sigmoidf_(gr + ar[i] + br_);
                const float z  = sigmoidf_(gz + az[i] + bz_);
                const float nn = tanhf(gn + r * (an[i] + bn_));
                const float hold = (float)hsh[brow * HSTRIDE + col];
                hstage[c][i] = (t < lsh[brow]) ? ((1.0f - z) * nn + z * hold)
                                               : hold;
            }
        }
        __syncthreads();  // all reads of h done
#pragma unroll
        for (int c = 0; c < 4; ++c) {
            const int combo = wave * 4 + c;
            const int m0 = (combo >> 4) * 16;
            const int col = (combo & 15) * 16 + nlo;
#pragma unroll
            for (int i = 0; i < 8; ++i) {
                const int brow = m0 + i + (laneHi << 3);
                const __bf16 hb = (__bf16)hstage[c][i];
                hsh[brow * HSTRIDE + col] = hb;
                if (writeOut)
                    outBuf[((size_t)brow * Tv + t) * (2 * Hv) + outOff + col] = hb;
            }
        }
        __syncthreads();  // h writes visible
    }
    // final hidden -> f32 global
    for (int i = threadIdx.x; i < Bv * Hv; i += 1024) {
        const int b = i >> 8, j = i & (Hv - 1);
        hFin[i] = (float)hsh[b * HSTRIDE + j];
    }
}

// ---------------------------------------------------------------------------
// out[b] = hf[b,:]·w_out[0:256] + hb[b,:]·w_out[256:512] + b_out
// ---------------------------------------------------------------------------
__global__ void final_proj_kernel(const float* __restrict__ hF,
                                  const float* __restrict__ hB,
                                  const float* __restrict__ wout,
                                  const float* __restrict__ bout,
                                  float* __restrict__ out) {
    const int b = threadIdx.x;  // 128 threads
    float acc = bout[0];
    for (int j = 0; j < Hv; ++j)
        acc += hF[b * Hv + j] * wout[j] + hB[b * Hv + j] * wout[Hv + j];
    out[b] = acc;
}

// ---------------------------------------------------------------------------
extern "C" void kernel_launch(void* const* d_in, const int* in_sizes, int n_in,
                              void* d_out, int out_size, void* d_ws, size_t ws_size,
                              hipStream_t stream) {
    const int*   x     = (const int*)  d_in[0];
    const int*   len   = (const int*)  d_in[1];
    const float* emb   = (const float*)d_in[2];
    const float* Wih0f = (const float*)d_in[3];
    const float* Whh0f = (const float*)d_in[4];
    const float* bih0f = (const float*)d_in[5];
    const float* bhh0f = (const float*)d_in[6];
    const float* Wih0b = (const float*)d_in[7];
    const float* Whh0b = (const float*)d_in[8];
    const float* bih0b = (const float*)d_in[9];
    const float* bhh0b = (const float*)d_in[10];
    const float* Wih1f = (const float*)d_in[11];
    const float* Whh1f = (const float*)d_in[12];
    const float* bih1f = (const float*)d_in[13];
    const float* bhh1f = (const float*)d_in[14];
    const float* Wih1b = (const float*)d_in[15];
    const float* Whh1b = (const float*)d_in[16];
    const float* bih1b = (const float*)d_in[17];
    const float* bhh1b = (const float*)d_in[18];
    const float* wout  = (const float*)d_in[19];
    const float* bout  = (const float*)d_in[20];
    float* out = (float*)d_out;

    // ---- workspace carve (bump allocator, 256B aligned) ----
    char* ws = (char*)d_ws;
    size_t off = 0;
    auto alloc = [&](size_t bytes) -> char* {
        char* p = ws + off;
        off += (bytes + 255) & ~(size_t)255;
        return p;
    };
    __bf16* X0  = (__bf16*)alloc((size_t)MROWS * Ev * 2);       //  33.6 MB
    __bf16* X1  = (__bf16*)alloc((size_t)MROWS * 2 * Hv * 2);   // 134.2 MB
    __bf16* GXf = (__bf16*)alloc((size_t)MROWS * G3H * 2);      // 201.3 MB
    __bf16* GXb = (__bf16*)alloc((size_t)MROWS * G3H * 2);      // 201.3 MB
    __bf16* wih0f = (__bf16*)alloc((size_t)G3H * Ev * 2);
    __bf16* wih0b = (__bf16*)alloc((size_t)G3H * Ev * 2);
    __bf16* whh0f = (__bf16*)alloc((size_t)G3H * Hv * 2);
    __bf16* whh0b = (__bf16*)alloc((size_t)G3H * Hv * 2);
    __bf16* wih1f = (__bf16*)alloc((size_t)G3H * 2 * Hv * 2);
    __bf16* wih1b = (__bf16*)alloc((size_t)G3H * 2 * Hv * 2);
    __bf16* whh1f = (__bf16*)alloc((size_t)G3H * Hv * 2);
    __bf16* whh1b = (__bf16*)alloc((size_t)G3H * Hv * 2);
    float*  hF = (float*)alloc((size_t)Bv * Hv * 4);
    float*  hB = (float*)alloc((size_t)Bv * Hv * 4);

    auto cvt = [&](const float* src, __bf16* dst, int n) {
        f32_to_bf16_kernel<<<(n + 255) / 256, 256, 0, stream>>>(src, dst, n);
    };
    // ---- 1. weights -> bf16 ----
    cvt(Wih0f, wih0f, G3H * Ev);
    cvt(Wih0b, wih0b, G3H * Ev);
    cvt(Whh0f, whh0f, G3H * Hv);
    cvt(Whh0b, whh0b, G3H * Hv);
    cvt(Wih1f, wih1f, G3H * 2 * Hv);
    cvt(Wih1b, wih1b, G3H * 2 * Hv);
    cvt(Whh1f, whh1f, G3H * Hv);
    cvt(Whh1b, whh1b, G3H * Hv);

    // ---- 2. embedding gather ----
    {
        const size_t tot = (size_t)MROWS * (Ev / 4);
        embed_bf16_kernel<<<(unsigned)((tot + 255) / 256), 256, 0, stream>>>(x, emb, X0);
    }

    const dim3 gGrid(MROWS / 128, G3H / 96);  // 1024 x 8
    const size_t scanShm = (size_t)Bv * (Hv + 8) * 2 + Bv * 4;  // 68.1 KB

    // ---- 3. layer 0 ----
    gemm_gates_bf16<<<gGrid, 256, 0, stream>>>(X0, wih0f, bih0f, GXf, Ev);
    gemm_gates_bf16<<<gGrid, 256, 0, stream>>>(X0, wih0b, bih0b, GXb, Ev);
    gru_scan_kernel<<<2, 1024, scanShm, stream>>>(
        GXf, GXb, whh0f, whh0b, bhh0f, bhh0b, len, X1, /*writeOut=*/1, hF, hB);

    // ---- 4. layer 1 (only final hiddens needed) ----
    gemm_gates_bf16<<<gGrid, 256, 0, stream>>>(X1, wih1f, bih1f, GXf, 2 * Hv);
    gemm_gates_bf16<<<gGrid, 256, 0, stream>>>(X1, wih1b, bih1b, GXb, 2 * Hv);
    gru_scan_kernel<<<2, 1024, scanShm, stream>>>(
        GXf, GXb, whh1f, whh1b, bhh1f, bhh1b, len, X1, /*writeOut=*/0, hF, hB);

    // ---- 5. output projection ----
    final_proj_kernel<<<1, 128, 0, stream>>>(hF, hB, wout, bout, out);
}
